// ChebyshevKANLayer_13898514170640
// MI455X (gfx1250) — matrix operand order repaired and checked by reading, not compiled, not run on hardware
//
#include <hip/hip_runtime.h>

// ---------------------------------------------------------------------------
// Chebyshev KAN layer as a bf16 WMMA GEMM on gfx1250 (wave32).
//   y[b,o] = sum_{i,j} T_j(xhat[b,i]) * C[i,o,j]   (M=8192,N=1024,K=9216)
// B tile staged with GLOBAL_LOAD_ASYNC_TO_LDS_B128 (ASYNCcnt), overlapped
// with on-the-fly Chebyshev A-tile generation.
// ---------------------------------------------------------------------------

typedef __attribute__((ext_vector_type(16))) __bf16 bf16x16;
typedef __attribute__((ext_vector_type(8)))  float  floatx8;

#define BATCH    8192
#define IN_DIM   1024
#define OUT_DIM  1024
#define DEG      9              // degree+1 Chebyshev terms
#define CI       32             // i-columns per K-chunk
#define KC       (CI * DEG)     // 288 K-values per chunk
#define NCHUNK   (IN_DIM / CI)  // 32 chunks
#define KTOT     (IN_DIM * DEG) // 9216
#define BM       128
#define BN       128
#define LDK      296            // padded LDS K-stride (breaks 64-bank conflicts)

// -------------------------- kernel 1: row min/max --------------------------
__global__ __launch_bounds__(256) void rowminmax_kernel(
    const float* __restrict__ x, float* __restrict__ rs, float* __restrict__ rc) {
  __shared__ float smn[256];
  __shared__ float smx[256];
  const int tid = threadIdx.x;
  const int row = blockIdx.x;
  const float4 v = ((const float4*)(x + (size_t)row * IN_DIM))[tid];
  float mn = fminf(fminf(v.x, v.y), fminf(v.z, v.w));
  float mx = fmaxf(fmaxf(v.x, v.y), fmaxf(v.z, v.w));
  smn[tid] = mn; smx[tid] = mx;
  __syncthreads();
  for (int s = 128; s > 0; s >>= 1) {
    if (tid < s) {
      smn[tid] = fminf(smn[tid], smn[tid + s]);
      smx[tid] = fmaxf(smx[tid], smx[tid + s]);
    }
    __syncthreads();
  }
  if (tid == 0) {
    const float s2 = 2.0f / (smx[0] - smn[0]);
    rs[row] = s2;                  // xhat = s2*x + c
    rc[row] = -s2 * smn[0] - 1.0f;
  }
}

// ------------------ kernel 2: repack coeffs -> bf16 B^T --------------------
// Bt[o][k] with k = chunk*288 + d*32 + ii  (i = chunk*32 + ii)
__global__ __launch_bounds__(256) void convB_kernel(
    const float* __restrict__ coeffs, __bf16* __restrict__ Bt) {
  const size_t idx = (size_t)blockIdx.x * 256 + threadIdx.x; // 1024*9216 total
  const int o  = (int)(idx / KTOT);
  const int k  = (int)(idx % KTOT);
  const int ch = k / KC;
  const int r  = k % KC;
  const int d  = r / CI;
  const int ii = r % CI;
  const int i  = ch * CI + ii;
  Bt[idx] = (__bf16)coeffs[((size_t)i * OUT_DIM + o) * DEG + d];
}

// --------------------------- kernel 3: WMMA GEMM ---------------------------
__global__ __launch_bounds__(256, 1) void cheby_gemm_kernel(
    const float* __restrict__ x, const __bf16* __restrict__ Bt,
    const float* __restrict__ rs, const float* __restrict__ rc,
    float* __restrict__ y) {
  __shared__ __bf16 As[BM][LDK];   // A tile, m-major, k contiguous
  __shared__ __bf16 Bs[BN][LDK];   // B tile, n-major (transposed), k contiguous

  const int tid = threadIdx.x;
  const int bm  = blockIdx.y * BM;
  const int bn  = blockIdx.x * BN;

  const int lane = tid & 31;
  const int wid  = tid >> 5;           // 8 waves
  const int wm   = (wid & 3) * 32;     // wave row offset within 128
  const int wn   = (wid >> 2) * 64;    // wave col offset within 128
  const int l15  = lane & 15;
  const int lh   = lane >> 4;          // half-wave select

  floatx8 acc[2][4];
#pragma unroll
  for (int mi = 0; mi < 2; ++mi)
#pragma unroll
    for (int ni = 0; ni < 4; ++ni)
#pragma unroll
      for (int e = 0; e < 8; ++e) acc[mi][ni][e] = 0.0f;

  // cooperative staging mapping: 2 threads per row, 128 rows
  const int sm = tid >> 1;
  const int sh = tid & 1;
  const float srow = rs[bm + sm];
  const float crow = rc[bm + sm];
  const float*  xrow = x  + (size_t)(bm + sm) * IN_DIM + sh * 16;
  const __bf16* brow = Bt + (size_t)(bn + sm) * KTOT   + sh * (KC / 2);

  // LDS byte offset for this thread's B destination (low 32 bits of generic
  // LDS address == LDS offset on gfx1250 flat aperture).
  const unsigned bdstOff =
      (unsigned)(unsigned long long)(uintptr_t)(&Bs[sm][sh * (KC / 2)]);

  for (int ch = 0; ch < NCHUNK; ++ch) {
    __syncthreads();  // previous iteration's WMMAs done before overwrite

    // ---- stage B: 18 x async b128 DMA, global -> LDS (ASYNCcnt) ----
    {
      const unsigned long long gsrc =
          (unsigned long long)(uintptr_t)(brow + (size_t)ch * KC);
#pragma unroll
      for (int t = 0; t < 18; ++t) {
        asm volatile("global_load_async_to_lds_b128 %0, %1, off offset:%2"
                     :: "v"(bdstOff), "v"(gsrc), "i"(t * 16)
                     : "memory");
      }
    }

    // ---- stage A (overlaps async B): normalize, Chebyshev, bf16 ----
    {
      float4 xv[4];
      const float4* xs = (const float4*)(xrow + (size_t)ch * CI);
#pragma unroll
      for (int t = 0; t < 4; ++t) xv[t] = xs[t];
      const float* xf = (const float*)xv;
#pragma unroll
      for (int g = 0; g < 2; ++g) {  // two groups of 8 to cap VGPR pressure
        alignas(16) __bf16 tv[DEG][8];
#pragma unroll
        for (int q = 0; q < 8; ++q) {
          const float xn = xf[g * 8 + q] * srow + crow;
          float t0 = 1.0f, t1 = xn;
          tv[0][q] = (__bf16)t0;
          tv[1][q] = (__bf16)t1;
#pragma unroll
          for (int d = 2; d < DEG; ++d) {
            const float t2 = 2.0f * xn * t1 - t0;
            tv[d][q] = (__bf16)t2;
            t0 = t1; t1 = t2;
          }
        }
#pragma unroll
        for (int d = 0; d < DEG; ++d)
          *(float4*)(&As[sm][d * CI + sh * 16 + g * 8]) = *(const float4*)(&tv[d][0]);
      }
    }

    if (ch + 1 < NCHUNK) {  // global_prefetch next chunk
      __builtin_prefetch(brow + (size_t)(ch + 1) * KC, 0, 3);
      __builtin_prefetch(xrow + (size_t)(ch + 1) * CI, 0, 3);
    }

    asm volatile("s_wait_asynccnt 0x0" ::: "memory");  // B tile landed
    __syncthreads();                                   // tiles visible to all

    // ---- 9 k-steps of 32; 8 WMMAs each (2 M-subtiles x 4 N-subtiles) ----
#pragma unroll
    for (int ks = 0; ks < DEG; ++ks) {
      bf16x16 af[2], bf[4];
#pragma unroll
      for (int mi = 0; mi < 2; ++mi) {
        // A 16x32 bf16: lanes 0-15 K{0-7,16-23}, lanes 16-31 K{8-15,24-31}
        ((float4*)&af[mi])[0] = *(const float4*)(&As[wm + mi * 16 + l15][ks * 32 + lh * 8]);
        ((float4*)&af[mi])[1] = *(const float4*)(&As[wm + mi * 16 + l15][ks * 32 + 16 + lh * 8]);
      }
#pragma unroll
      for (int ni = 0; ni < 4; ++ni) {
        // B 32x16 bf16: lane's column = l15, lanes 0-15 K{0-7,16-23}, lanes 16-31 K{8-15,24-31}
        ((float4*)&bf[ni])[0] = *(const float4*)(&Bs[wn + ni * 16 + l15][ks * 32 + lh * 8]);
        ((float4*)&bf[ni])[1] = *(const float4*)(&Bs[wn + ni * 16 + l15][ks * 32 + 16 + lh * 8]);
      }
#pragma unroll
      for (int mi = 0; mi < 2; ++mi)
#pragma unroll
        for (int ni = 0; ni < 4; ++ni)
          acc[mi][ni] = __builtin_amdgcn_wmma_f32_16x16x32_bf16(
              false, af[mi], false, bf[ni], (short)0, acc[mi][ni], false, false);
    }
  }

  // ---- epilogue: C/D layout VGPR r -> M = r + lh*8, N = l15 ----
#pragma unroll
  for (int mi = 0; mi < 2; ++mi)
#pragma unroll
    for (int ni = 0; ni < 4; ++ni) {
      const int row0 = bm + wm + mi * 16 + lh * 8;
      const int col  = bn + wn + ni * 16 + l15;
#pragma unroll
      for (int r = 0; r < 8; ++r)
        y[(size_t)(row0 + r) * OUT_DIM + col] = acc[mi][ni][r];
    }
}

// ------------------------------- launcher ----------------------------------
extern "C" void kernel_launch(void* const* d_in, const int* in_sizes, int n_in,
                              void* d_out, int out_size, void* d_ws, size_t ws_size,
                              hipStream_t stream) {
  (void)in_sizes; (void)n_in; (void)out_size; (void)ws_size;
  const float* x      = (const float*)d_in[0];   // [8192,1024] f32
  const float* coeffs = (const float*)d_in[1];   // [1024,1024,9] f32
  float* y = (float*)d_out;                      // [8192,1024] f32

  float*  rs = (float*)d_ws;                           // [8192] scale
  float*  rc = rs + BATCH;                             // [8192] offset
  __bf16* Bt = (__bf16*)((char*)d_ws + 65536);         // [1024][9216] bf16 (~18MB)

  rowminmax_kernel<<<BATCH, 256, 0, stream>>>(x, rs, rc);

  const int convBlocks = (int)(((size_t)OUT_DIM * KTOT) / 256);  // 36864
  convB_kernel<<<convBlocks, 256, 0, stream>>>(coeffs, Bt);

  dim3 grid(OUT_DIM / BN, BATCH / BM);  // (8, 64)
  cheby_gemm_kernel<<<grid, 256, 0, stream>>>(x, Bt, rs, rc, y);
}
